// ReverseDiffusionDense_67327907332467
// MI455X (gfx1250) — matrix operand order repaired and checked by reading, not compiled, not run on hardware
//
#include <hip/hip_runtime.h>
#include <stdint.h>

// ---------- constants ----------
#define BV 2
#define NV 256
#define HIDV 256
#define EEDV 128
#define NODES (BV*NV)            // 512
#define EDGES ((size_t)BV*NV*NV) // 131072
#define PI_F 3.14159265358979f

// ---------- WMMA types ----------
typedef __attribute__((ext_vector_type(16))) __bf16 v16bf;
typedef __attribute__((ext_vector_type(8)))  float  v8f;

union FragU { v16bf v; unsigned int u[8]; };

// ---------- helpers ----------
__device__ __forceinline__ unsigned short f2bf(float f) {
    unsigned int u = __builtin_bit_cast(unsigned int, f);
    unsigned int r = u + 0x7FFFu + ((u >> 16) & 1u);   // RNE
    return (unsigned short)(r >> 16);
}
__device__ __forceinline__ float bf2f(unsigned int h) {
    return __builtin_bit_cast(float, h << 16);
}
__device__ __forceinline__ float silu(float x) { return x / (1.0f + __expf(-x)); }

// ---------- generic bf16 WMMA GEMM ----------
enum { GM_F32 = 0, GM_BIAS = 1, GM_EDGEPRE = 2, GM_MSILU = 3, GM_PHI = 4, GM_ADDSILU = 5, GM_RESID = 6 };

struct GemmArgs {
    const unsigned short* A;   // [M][K] bf16 row-major
    const unsigned short* Bt;  // [N][K] bf16 (i.e. weight transposed)
    int M, N, K, mode;
    const float* bias;         // [N]
    float* outF;               // [M][N]
    unsigned short* outB;      // [M][N] bf16
    const float* hi;           // [NODES][HID]   (EDGEPRE)
    const float* hj;           // [NODES][HID]   (EDGEPRE)
    const float* radial;       // [M]            (EDGEPRE)
    const float* wr;           // [N]  (EDGEPRE: Wr row; PHI: coord_w2)
    const float* mask;         // [NODES]        (MSILU)
    float* agg;                // MSILU: [NODES][HID] agg atomics; PHI: [M] phi atomics
    const float* addsrc;       // [M][N]         (ADDSILU)
};

// Block tile: 128 rows (8 waves x 16) x 128 cols, K = KT (compile-time).
// B tile (identical for all 8 waves) is staged once in LDS; WMMA B fragments
// are double-buffered from ds_load_b128, A fragments double-buffered from global.
template <int MODE, int KT>
__global__ __launch_bounds__(256) void k_gemm(GemmArgs g) {
    __shared__ unsigned short sB[128 * KT];   // 64KB (KT=256) / 32KB (KT=128)

    const int wave = threadIdx.x >> 5;
    const int lane = threadIdx.x & 31;
    const int hsel = lane >> 4;       // 0: lanes 0-15, 1: lanes 16-31
    const int lrow = lane & 15;
    const int m0 = (blockIdx.x * 8 + wave) * 16;   // M is always a multiple of 128
    const int n0 = blockIdx.y * 128;

    // ---- cooperative stage of B tile [n0, n0+128) x KT ----
    {
        const uint4* gsrc = (const uint4*)(g.Bt + (size_t)n0 * KT);
        uint4* ldst = (uint4*)sB;
        const int iters = (128 * KT / 8) / 256;   // 16 (KT=256) / 8 (KT=128)
#pragma unroll
        for (int t = 0; t < iters; ++t)
            ldst[t * 256 + threadIdx.x] = gsrc[t * 256 + threadIdx.x];
    }
    __syncthreads();

    v8f acc[8];
    const v8f vz = {0.f,0.f,0.f,0.f,0.f,0.f,0.f,0.f};
#pragma unroll
    for (int i = 0; i < 8; ++i) acc[i] = vz;

    const unsigned int* arow =
        (const unsigned int*)(g.A + (size_t)(m0 + lrow) * KT) + hsel * 4;

    const unsigned int* bl = (const unsigned int*)sB;
    int boff[8];
#pragma unroll
    for (int nt = 0; nt < 8; ++nt)
        boff[nt] = ((nt * 16 + lrow) * KT >> 1) + hsel * 4;

    const int KI = KT / 32;
    FragU fa[2], fb[2];
#pragma unroll
    for (int v = 0; v < 4; ++v) { fa[0].u[v] = arow[v]; fa[0].u[4 + v] = arow[8 + v]; }

#pragma unroll
    for (int ki = 0; ki < KI; ++ki) {
        const int kd = ki * 16;   // dword offset within row
        if (ki + 1 < KI) {        // prefetch next A fragment (covered by 8 WMMAs)
            const int kn = kd + 16;
#pragma unroll
            for (int v = 0; v < 4; ++v) {
                fa[(ki + 1) & 1].u[v] = arow[kn + v];
                fa[(ki + 1) & 1].u[4 + v] = arow[kn + 8 + v];
            }
        }
#pragma unroll
        for (int v = 0; v < 8; ++v) fb[0].u[v] = bl[boff[0] + kd + (v & 3) + ((v >> 2) << 3)];
#pragma unroll
        for (int nt = 0; nt < 8; ++nt) {
            if (nt < 7) {         // issue next B fragment before consuming current
#pragma unroll
                for (int v = 0; v < 8; ++v) fb[(nt + 1) & 1].u[v] = bl[boff[nt + 1] + kd + (v & 3) + ((v >> 2) << 3)];
            }
            acc[nt] = __builtin_amdgcn_wmma_f32_16x16x32_bf16(
                false, fa[ki & 1].v, false, fb[nt & 1].v, (short)0, acc[nt], false, false);
        }
    }

    // ---- epilogue ----
    const int Nn = g.N;
#pragma unroll
    for (int nt = 0; nt < 8; ++nt) {
        const int n = n0 + nt * 16 + lrow;
        float colsum = 0.0f;
#pragma unroll
        for (int r = 0; r < 8; ++r) {
            const int m = m0 + r + hsel * 8;
            const size_t o = (size_t)m * Nn + n;
            float v = acc[nt][r];
            if (MODE == GM_F32) {
                g.outF[o] = v;
            } else if (MODE == GM_BIAS) {
                v += g.bias[n];
                if (g.outF) g.outF[o] = v;
                if (g.outB) g.outB[o] = f2bf(v);
            } else if (MODE == GM_EDGEPRE) {
                const int node_i = m >> 8;
                const int b = m >> 16;
                const int node_j = (b << 8) | (m & 255);
                v += g.hi[(size_t)node_i * HIDV + n] + g.hj[(size_t)node_j * HIDV + n] +
                     g.radial[m] * g.wr[n] + g.bias[n];
                g.outB[o] = f2bf(silu(v));
            } else if (MODE == GM_MSILU) {
                v = silu(v + g.bias[n]);
                const int node_i = m >> 8;
                const int b = m >> 16;
                const int jj = m & 255;
                const int ii = node_i & 255;
                const float em = (ii == jj) ? 0.0f
                               : g.mask[node_i] * g.mask[(b << 8) | jj];
                v *= em;
                g.outB[o] = f2bf(v);
                colsum += v;
            } else if (MODE == GM_PHI) {
                // phi[m] += sum_n silu(acc + cb1[n]) * cw2[n]   (t2 never materialized)
                v = silu(v + g.bias[n]) * g.wr[n];
#pragma unroll
                for (int off = 1; off < 16; off <<= 1) v += __shfl_xor(v, off, 32);
                if (lrow == 0) atomicAdd(&g.agg[m], v);
            } else if (MODE == GM_ADDSILU) {
                v += g.addsrc[o] + g.bias[n];
                g.outB[o] = f2bf(silu(v));
            } else { // GM_RESID
                g.outF[o] += v + g.bias[n];
            }
        }
        if (MODE == GM_MSILU) {
            // rows m0..m0+15 share node_i; combine both half-rows then one atomic per column
            colsum += __shfl_xor(colsum, 16, 32);
            if (hsel == 0)
                atomicAdd(&g.agg[(size_t)(m0 >> 8) * HIDV + n], colsum);
        }
    }
}

// ---------- elementwise / reduction kernels ----------
__global__ void k_transpose(const float* src, unsigned short* dst,
                            int Kdim, int Ndim, int row_off, int ld) {
    int idx = blockIdx.x * 256 + threadIdx.x;
    if (idx >= Kdim * Ndim) return;
    int k = idx % Kdim, n = idx / Kdim;
    dst[(size_t)n * Kdim + k] = f2bf(src[(size_t)(row_off + k) * ld + n]);
}

__global__ void k_fill0(float* p, int n) {
    int i = blockIdx.x * 256 + threadIdx.x;
    if (i < n) p[i] = 0.0f;
}

__global__ void k_copy(const float* s, float* d, int n) {
    int i = blockIdx.x * 256 + threadIdx.x;
    if (i < n) d[i] = s[i];
}

__global__ void k_f2bf(const float* s, unsigned short* d, int n) {
    int i = blockIdx.x * 256 + threadIdx.x;
    if (i < n) d[i] = f2bf(s[i]);
}

__global__ void k_h0(const int* ridx, const int* tarr, const float* pet, float* h) {
    int idx = blockIdx.x * 256 + threadIdx.x;   // NODES*HID
    int c = idx & 255, node = idx >> 8, b = node >> 8;
    float id = (float)ridx[node];
    int Kc = c & 127;
    float ang = id * PI_F / __powf(256.0f, 2.0f * (float)Kc / 256.0f);
    float e = (c < 128) ? __sinf(ang) : __cosf(ang);
    h[idx] = e + pet[(size_t)tarr[b] * 256 + c];
}

__global__ void k_edgeattr(const int* ridx, const float* mask, unsigned short* ea) {
    size_t idx = (size_t)blockIdx.x * 256 + threadIdx.x;  // EDGES*EED
    int k = (int)(idx & 127);
    size_t m = idx >> 7;
    int node_i = (int)(m >> 8);
    int b = (int)(m >> 16);
    int node_j = (b << 8) | (int)(m & 255);
    float rel = (float)(ridx[node_i] - ridx[node_j]);
    int Kc = k & 63;
    float ang = rel * PI_F / __powf(256.0f, 2.0f * (float)Kc / 128.0f);
    float v = (k < 64) ? __sinf(ang) : __cosf(ang);
    ea[idx] = f2bf(v * mask[node_i] * mask[node_j]);
}

__global__ void k_radial(const float* x, float* radial, float* ndiff) {
    size_t m = (size_t)blockIdx.x * 256 + threadIdx.x;    // EDGES
    int node_i = (int)(m >> 8);
    int b = (int)(m >> 16);
    int node_j = (b << 8) | (int)(m & 255);
    float dx = x[node_i * 3 + 0] - x[node_j * 3 + 0];
    float dy = x[node_i * 3 + 1] - x[node_j * 3 + 1];
    float dz = x[node_i * 3 + 2] - x[node_j * 3 + 2];
    float r = dx * dx + dy * dy + dz * dz;
    float inv = 1.0f / (sqrtf(r) + 1e-8f);
    radial[m] = r;
    ndiff[m * 3 + 0] = dx * inv;
    ndiff[m * 3 + 1] = dy * inv;
    ndiff[m * 3 + 2] = dz * inv;
}

__global__ void k_maskh(float* h, unsigned short* hb, const float* mask) {
    int e = blockIdx.x * 256 + threadIdx.x;    // NODES*HID
    float v = h[e] * mask[e >> 8];
    h[e] = v;
    hb[e] = f2bf(v);
}

__global__ void k_maskx(float* x, const float* mask, int n) {
    int e = blockIdx.x * 256 + threadIdx.x;
    if (e < n) x[e] *= mask[e / 3];
}

// x update: one workgroup per node (b,i), 256 threads over j
__global__ void k_xupdate(const float* ndiff, const float* phi, const float* mask, float* x) {
    int node = blockIdx.x;                 // 0..NODES-1
    int j = threadIdx.x;
    int b = node >> 8, i = node & 255;
    size_t m = (size_t)node * 256 + j;
    float em = (i == j) ? 0.0f : mask[node] * mask[(b << 8) | j];
    float ph = phi[m] * em;
    __shared__ float red[4][256];
    red[0][j] = ndiff[m * 3 + 0] * ph;
    red[1][j] = ndiff[m * 3 + 1] * ph;
    red[2][j] = ndiff[m * 3 + 2] * ph;
    red[3][j] = em;
    __syncthreads();
    for (int s = 128; s > 0; s >>= 1) {
        if (j < s) {
#pragma unroll
            for (int d = 0; d < 4; ++d) red[d][j] += red[d][j + s];
        }
        __syncthreads();
    }
    if (j == 0) {
        float den = fmaxf(red[3][0], 1.0f);
        float mi = mask[node];
        for (int d = 0; d < 3; ++d)
            x[node * 3 + d] = (x[node * 3 + d] + red[d][0] / den) * mi;
    }
}

__global__ void k_layernorm(const float* hin, const float* mask,
                            const float* gw, const float* bw, float* hout) {
    int row = blockIdx.x, c = threadIdx.x;
    float v = hin[(size_t)row * 256 + c] * mask[row];
    __shared__ float red[256];
    red[c] = v; __syncthreads();
    for (int s = 128; s > 0; s >>= 1) { if (c < s) red[c] += red[c + s]; __syncthreads(); }
    float mu = red[0] * (1.0f / 256.0f);
    __syncthreads();
    float d = v - mu;
    red[c] = d * d; __syncthreads();
    for (int s = 128; s > 0; s >>= 1) { if (c < s) red[c] += red[c + s]; __syncthreads(); }
    float var = red[0] * (1.0f / 256.0f);
    hout[(size_t)row * 256 + c] = d * rsqrtf(var + 1e-5f) * gw[c] + bw[c];
}

__global__ void k_out(const float* x, const float* bb, const float* mask, float* out, int n) {
    int e = blockIdx.x * 256 + threadIdx.x;
    if (e < n) {
        float mi = mask[e / 3];
        out[e] = (x[e] - bb[e] * mi) * mi;
    }
}

// ---------- host orchestration ----------
extern "C" void kernel_launch(void* const* d_in, const int* in_sizes, int n_in,
                              void* d_out, int out_size, void* d_ws, size_t ws_size,
                              hipStream_t stream) {
    (void)in_sizes; (void)n_in; (void)out_size; (void)ws_size;
    const float* bb   = (const float*)d_in[0];
    const float* mask = (const float*)d_in[1];
    const int*   ridx = (const int*)d_in[2];
    const int*   tarr = (const int*)d_in[3];
    const float* pet  = (const float*)d_in[4];
    float* out = (float*)d_out;

    auto leaf = [&](int i) { return (const float*)d_in[i]; };
    // params pytree (sorted-key flatten): per layer (50 leaves):
    //  +0 bin, +1 bout, gcl[s] at +2+11*s:
    //    +0 coord_b1, +1 coord_w1, +2 coord_w2, +3 edge_b1, +4 edge_b2,
    //    +5 edge_w1, +6 edge_w2, +7 node_b1, +8 node_b2, +9 node_w1, +10 node_w2
    //  +46 ln_b, +47 ln_g, +48 win, +49 wout

    // ---- carve workspace ----
    char* p = (char*)d_ws;
    auto alloc = [&](size_t bytes) -> char* {
        char* r = p; p += (bytes + 255) & ~(size_t)255; return r;
    };
    unsigned short* EA   = (unsigned short*)alloc(EDGES * EEDV * 2);
    unsigned short* preB = (unsigned short*)alloc(EDGES * HIDV * 2);
    unsigned short* mB   = (unsigned short*)alloc(EDGES * HIDV * 2);
    float* hF    = (float*)alloc((size_t)NODES * HIDV * 4);
    unsigned short* hB   = (unsigned short*)alloc((size_t)NODES * HIDV * 2);
    float* hgF   = (float*)alloc((size_t)NODES * HIDV * 4);
    unsigned short* hgB  = (unsigned short*)alloc((size_t)NODES * HIDV * 2);
    float* hiF   = (float*)alloc((size_t)NODES * HIDV * 4);
    float* hjF   = (float*)alloc((size_t)NODES * HIDV * 4);
    float* radial= (float*)alloc(EDGES * 4);
    float* ndiff = (float*)alloc(EDGES * 3 * 4);
    float* phi   = (float*)alloc(EDGES * 4);
    float* agg   = (float*)alloc((size_t)NODES * HIDV * 4);
    unsigned short* aggB = (unsigned short*)alloc((size_t)NODES * HIDV * 2);
    float* tmp1  = (float*)alloc((size_t)NODES * HIDV * 4);
    unsigned short* pnB  = (unsigned short*)alloc((size_t)NODES * HIDV * 2);
    float* xF    = (float*)alloc((size_t)NODES * 3 * 4);
    float* houtF = (float*)alloc((size_t)NODES * HIDV * 4);
    unsigned short* wts = (unsigned short*)alloc((size_t)4 * 2097152 * 2);

    auto xpose = [&](const float* src, unsigned short* dst, int Kd, int Nd, int roff, int ld) {
        int total = Kd * Nd;
        k_transpose<<<(total + 255) / 256, 256, 0, stream>>>(src, dst, Kd, Nd, roff, ld);
    };
    auto gemm = [&](GemmArgs ga) {
        dim3 grid(ga.M / 128, ga.N / 128);
        if (ga.mode == GM_EDGEPRE) {
            k_gemm<GM_EDGEPRE, 128><<<grid, 256, 0, stream>>>(ga);
            return;
        }
        switch (ga.mode) {
        case GM_F32:     k_gemm<GM_F32, 256><<<grid, 256, 0, stream>>>(ga); break;
        case GM_BIAS:    k_gemm<GM_BIAS, 256><<<grid, 256, 0, stream>>>(ga); break;
        case GM_MSILU:   k_gemm<GM_MSILU, 256><<<grid, 256, 0, stream>>>(ga); break;
        case GM_PHI:     k_gemm<GM_PHI, 256><<<grid, 256, 0, stream>>>(ga); break;
        case GM_ADDSILU: k_gemm<GM_ADDSILU, 256><<<grid, 256, 0, stream>>>(ga); break;
        default:         k_gemm<GM_RESID, 256><<<grid, 256, 0, stream>>>(ga); break;
        }
    };

    // ---- convert all weights to bf16 transposed layout (every call; deterministic) ----
    for (int L = 0; L < 4; ++L) {
        int base = 5 + 50 * L;
        unsigned short* wl = wts + (size_t)L * 2097152;
        xpose(leaf(base + 48), wl, 256, 256, 0, 256);            // win_t
        xpose(leaf(base + 49), wl + 65536, 256, 256, 0, 256);    // wout_t
        for (int s = 0; s < 4; ++s) {
            int gb = base + 2 + 11 * s;
            unsigned short* wg = wl + 131072 + (size_t)s * 491520;
            const float* ew1 = leaf(gb + 5);
            xpose(ew1, wg,           256, 256, 0, 256);          // Wi_t
            xpose(ew1, wg + 65536,   256, 256, 256, 256);        // Wj_t
            xpose(ew1, wg + 131072,  128, 256, 513, 256);        // We_t
            xpose(leaf(gb + 6), wg + 163840, 256, 256, 0, 256);  // edge_w2_t
            xpose(leaf(gb + 1), wg + 229376, 256, 256, 0, 256);  // coord_w1_t
            const float* nw1 = leaf(gb + 9);
            xpose(nw1, wg + 294912, 256, 256, 0, 256);           // node_w1a_t
            xpose(nw1, wg + 360448, 256, 256, 256, 256);         // node_w1b_t
            xpose(leaf(gb + 10), wg + 425984, 256, 256, 0, 256); // node_w2_t
        }
    }

    // ---- init ----
    k_copy<<<(NODES * 3 + 255) / 256, 256, 0, stream>>>(bb, xF, NODES * 3);
    k_h0<<<NODES * HIDV / 256, 256, 0, stream>>>(ridx, tarr, pet, hF);
    k_edgeattr<<<(unsigned)(EDGES * EEDV / 256), 256, 0, stream>>>(ridx, mask, EA);

    for (int L = 0; L < 4; ++L) {
        int base = 5 + 50 * L;
        unsigned short* wl = wts + (size_t)L * 2097152;

        k_maskh<<<NODES * HIDV / 256, 256, 0, stream>>>(hF, hB, mask);
        k_maskx<<<(NODES * 3 + 255) / 256, 256, 0, stream>>>(xF, mask, NODES * 3);

        { GemmArgs ga = {}; ga.A = hB; ga.Bt = wl; ga.M = NODES; ga.N = 256; ga.K = 256;
          ga.mode = GM_BIAS; ga.bias = leaf(base + 0); ga.outF = hgF; ga.outB = hgB; gemm(ga); }

        for (int s = 0; s < 4; ++s) {
            int gb = base + 2 + 11 * s;
            unsigned short* wg = wl + 131072 + (size_t)s * 491520;
            const float* wr = leaf(gb + 5) + (size_t)512 * 256;   // edge_w1 row 512

            k_radial<<<(unsigned)(EDGES / 256), 256, 0, stream>>>(xF, radial, ndiff);

            { GemmArgs ga = {}; ga.A = hgB; ga.Bt = wg; ga.M = NODES; ga.N = 256; ga.K = 256;
              ga.mode = GM_F32; ga.outF = hiF; gemm(ga); }
            { GemmArgs ga = {}; ga.A = hgB; ga.Bt = wg + 65536; ga.M = NODES; ga.N = 256; ga.K = 256;
              ga.mode = GM_F32; ga.outF = hjF; gemm(ga); }

            { GemmArgs ga = {}; ga.A = EA; ga.Bt = wg + 131072; ga.M = (int)EDGES; ga.N = 256; ga.K = 128;
              ga.mode = GM_EDGEPRE; ga.bias = leaf(gb + 3); ga.hi = hiF; ga.hj = hjF;
              ga.radial = radial; ga.wr = wr; ga.outB = preB; gemm(ga); }

            k_fill0<<<NODES * HIDV / 256, 256, 0, stream>>>(agg, NODES * HIDV);

            { GemmArgs ga = {}; ga.A = preB; ga.Bt = wg + 163840; ga.M = (int)EDGES; ga.N = 256; ga.K = 256;
              ga.mode = GM_MSILU; ga.bias = leaf(gb + 4); ga.mask = mask; ga.agg = agg;
              ga.outB = mB; gemm(ga); }

            // phi = silu(m@coord_w1 + cb1) @ coord_w2, fused into the GEMM epilogue
            k_fill0<<<(unsigned)(EDGES / 256), 256, 0, stream>>>(phi, (int)EDGES);
            { GemmArgs ga = {}; ga.A = mB; ga.Bt = wg + 229376; ga.M = (int)EDGES; ga.N = 256; ga.K = 256;
              ga.mode = GM_PHI; ga.bias = leaf(gb + 0); ga.wr = leaf(gb + 2); ga.agg = phi; gemm(ga); }

            k_xupdate<<<NODES, 256, 0, stream>>>(ndiff, phi, mask, xF);

            k_f2bf<<<NODES * HIDV / 256, 256, 0, stream>>>(agg, aggB, NODES * HIDV);

            { GemmArgs ga = {}; ga.A = hgB; ga.Bt = wg + 294912; ga.M = NODES; ga.N = 256; ga.K = 256;
              ga.mode = GM_F32; ga.outF = tmp1; gemm(ga); }
            { GemmArgs ga = {}; ga.A = aggB; ga.Bt = wg + 360448; ga.M = NODES; ga.N = 256; ga.K = 256;
              ga.mode = GM_ADDSILU; ga.addsrc = tmp1; ga.bias = leaf(gb + 7); ga.outB = pnB; gemm(ga); }
            { GemmArgs ga = {}; ga.A = pnB; ga.Bt = wg + 425984; ga.M = NODES; ga.N = 256; ga.K = 256;
              ga.mode = GM_RESID; ga.bias = leaf(gb + 8); ga.outF = hgF; gemm(ga); }

            k_f2bf<<<NODES * HIDV / 256, 256, 0, stream>>>(hgF, hgB, NODES * HIDV);
        }

        { GemmArgs ga = {}; ga.A = hgB; ga.Bt = wl + 65536; ga.M = NODES; ga.N = 256; ga.K = 256;
          ga.mode = GM_BIAS; ga.bias = leaf(base + 1); ga.outF = houtF; gemm(ga); }

        k_layernorm<<<NODES, 256, 0, stream>>>(houtF, mask, leaf(base + 47), leaf(base + 46), hF);
    }

    k_out<<<(NODES * 3 + 255) / 256, 256, 0, stream>>>(xF, bb, mask, out, NODES * 3);
}
